// DocREModel_78305843741025
// MI455X (gfx1250) — matrix operand order repaired and checked by reading, not compiled, not run on hardware
//
#include <hip/hip_runtime.h>
#include <hip/hip_bf16.h>
#include <math.h>

typedef __attribute__((ext_vector_type(16))) _Float16 v16h;
typedef __attribute__((ext_vector_type(8)))  _Float16 v8h;
typedef __attribute__((ext_vector_type(8)))  float    v8f;

// Build a 16x32 f16 A fragment (contiguous K = 16h + s layout) from two contiguous loads.
static __device__ __forceinline__ v16h load_afrag(const _Float16* p)
{
    v8h a0 = *(const v8h*)(p);        // K = akb + 0..7
    v8h a1 = *(const v8h*)(p + 8);    // K = akb + 8..15
    return __builtin_shufflevector(a0, a1, 0, 1, 2, 3, 4, 5, 6, 7,
                                           8, 9, 10, 11, 12, 13, 14, 15);
}

// Store one 16x16 f32 accumulator tile with bias/activation, predicated edges.
static __device__ __forceinline__ void
store_tile(float* C, int ldc, _Float16* Ch, int ldch, const v8f& acc,
           int mbase, int nbase, int M, int N,
           const float* bias, int act, int lane)
{
    const int ccol = nbase + (lane & 15);
    if (ccol >= N) return;
    const int rbase = mbase + ((lane >> 4) << 3);   // lanes 16-31 -> M+8
    #pragma unroll
    for (int r = 0; r < 8; ++r) {
        int row = rbase + r;
        if (row < M) {
            float v = acc[r];
            if (bias) v += bias[ccol];
            if (act == 1)      v = fmaxf(v, 0.f);
            else if (act == 2) v = tanhf(v);
            if (C)  C [(long)row * ldc  + ccol] = v;
            if (Ch) Ch[(long)row * ldch + ccol] = (_Float16)v;
        }
    }
}

// ---------------------------------------------------------------------------
// WMMA f16 GEMM, 32x32 C block per wave (2x2 tiles, 4 independent WMMAs/step).
//   Ah: M x Kp row-major f16 (Kp % 32 == 0, zero padded)
//   BT: N x Kp row-major f16 (B transposed, zero padded)
//   C : optional f32 out (ldc); Ch: optional f16 out (ldch); bias per column.
// act: 0 none, 1 relu, 2 tanh.  ~16 FLOP/B fragment intensity, branch-free
// inner loop: 8 x global_load_b128 -> 4 x v_wmma_f32_16x16x32_f16.
// ---------------------------------------------------------------------------
__global__ void __launch_bounds__(128)
wmma_gemm_f16(const _Float16* __restrict__ Ah, int Kp, long sA,
              const _Float16* __restrict__ BT, long sB,
              float* __restrict__ C, int ldc, long sC,
              _Float16* __restrict__ Ch, int ldch, long sCh,
              int M, int N,
              const float* __restrict__ bias, int act)
{
    const int lane = threadIdx.x & 31;
    const int wave = threadIdx.x >> 5;
    const int bn = (N + 31) >> 5;
    const int bm = (M + 31) >> 5;
    const int t  = blockIdx.x * 4 + wave;
    const int bi = t / bn;
    const int bj = t - bi * bn;
    if (bi >= bm) return;                      // wave-uniform exit

    const int bb = blockIdx.z;
    const int m0 = bi << 5, n0 = bj << 5;
    const int Mc = M - 1, Nc = N - 1;
    int ar0 = m0 + (lane & 15);      if (ar0 > Mc) ar0 = Mc;
    int ar1 = m0 + 16 + (lane & 15); if (ar1 > Mc) ar1 = Mc;
    int bc0 = n0 + (lane & 15);      if (bc0 > Nc) bc0 = Nc;
    int bc1 = n0 + 16 + (lane & 15); if (bc1 > Nc) bc1 = Nc;
    const int akb = (lane >> 4) * 16;
    const int bkb = (lane >> 4) * 16;
    const _Float16* A0 = Ah + (long)bb * sA + (long)ar0 * Kp + akb;
    const _Float16* A1 = Ah + (long)bb * sA + (long)ar1 * Kp + akb;
    const _Float16* B0 = BT + (long)bb * sB + (long)bc0 * Kp + bkb;
    const _Float16* B1 = BT + (long)bb * sB + (long)bc1 * Kp + bkb;

    v8f acc00 = {}, acc01 = {}, acc10 = {}, acc11 = {};
    for (int k0 = 0; k0 < Kp; k0 += 32) {
        v16h av0 = load_afrag(A0 + k0);
        v16h av1 = load_afrag(A1 + k0);
        v16h bv0 = *(const v16h*)(B0 + k0);
        v16h bv1 = *(const v16h*)(B1 + k0);
        __builtin_prefetch(A0 + k0 + 64, 0, 3);
        __builtin_prefetch(A1 + k0 + 64, 0, 3);
        __builtin_prefetch(B0 + k0 + 64, 0, 3);
        __builtin_prefetch(B1 + k0 + 64, 0, 3);
        acc00 = __builtin_amdgcn_wmma_f32_16x16x32_f16(
                    false, av0, false, bv0, (short)0, acc00, false, false);
        acc01 = __builtin_amdgcn_wmma_f32_16x16x32_f16(
                    false, av0, false, bv1, (short)0, acc01, false, false);
        acc10 = __builtin_amdgcn_wmma_f32_16x16x32_f16(
                    false, av1, false, bv0, (short)0, acc10, false, false);
        acc11 = __builtin_amdgcn_wmma_f32_16x16x32_f16(
                    false, av1, false, bv1, (short)0, acc11, false, false);
    }

    float*    Cb  = C  ? C  + (long)bb * sC  : nullptr;
    _Float16* Chb = Ch ? Ch + (long)bb * sCh : nullptr;
    store_tile(Cb, ldc, Chb, ldch, acc00, m0,      n0,      M, N, bias, act, lane);
    store_tile(Cb, ldc, Chb, ldch, acc01, m0,      n0 + 16, M, N, bias, act, lane);
    store_tile(Cb, ldc, Chb, ldch, acc10, m0 + 16, n0,      M, N, bias, act, lane);
    store_tile(Cb, ldc, Chb, ldch, acc11, m0 + 16, n0 + 16, M, N, bias, act, lane);
}

// ---------------------------------------------------------------------------
// Implicit-im2col WMMA conv, 5x5 SAME, 22x22 spatial, zero-halo f16 tensors.
//   Xp: (B,Cin,26,26) halo=0   Wh: (Cout, Cin*25) f16   Yp: (B,Cout,26,26)
// 2x2 tile block per wave; weight frags contiguous, gathers unconditional.
// ---------------------------------------------------------------------------
__global__ void __launch_bounds__(128)
wmma_conv_f16(const _Float16* __restrict__ Xp,
              const _Float16* __restrict__ Wh,
              const float* __restrict__ bias,
              _Float16* __restrict__ Yp,
              int Cin, int Cout)
{
    const int lane = threadIdx.x & 31;
    const int wave = threadIdx.x >> 5;
    const int bn = 16;                          // ceil(484/32)
    const int t  = blockIdx.x * 4 + wave;
    const int bi = t / bn;
    const int bj = t - bi * bn;
    if (bi >= (Cout >> 5)) return;

    const int bb = blockIdx.z;
    const int K  = Cin * 25;                    // multiple of 32
    const int m0 = bi << 5, n0 = bj << 5;
    const int akb = (lane >> 4) * 16;
    const int bkb = (lane >> 4) * 16;
    const _Float16* A0 = Wh + (long)(m0 + (lane & 15)) * K + akb;
    const _Float16* A1 = A0 + 16L * K;
    int bc0 = n0 + (lane & 15);      if (bc0 > 483) bc0 = 483;
    int bc1 = n0 + 16 + (lane & 15); if (bc1 > 483) bc1 = 483;
    const int oy0 = bc0 / 22, ox0 = bc0 - oy0 * 22;
    const int oy1 = bc1 / 22, ox1 = bc1 - oy1 * 22;
    const _Float16* Xbase = Xp + (long)bb * Cin * 676;
    const _Float16* Xb0 = Xbase + oy0 * 26 + ox0;
    const _Float16* Xb1 = Xbase + oy1 * 26 + ox1;

    v8f acc00 = {}, acc01 = {}, acc10 = {}, acc11 = {};
    for (int k0 = 0; k0 < K; k0 += 32) {
        v16h av0 = load_afrag(A0 + k0);
        v16h av1 = load_afrag(A1 + k0);
        v16h bv0, bv1;
        #pragma unroll
        for (int h = 0; h < 16; ++h) {
            int k  = k0 + bkb + h;
            int ci = k / 25, rem = k - ci * 25;
            int ky = rem / 5, kx = rem - ky * 5;
            int off = ci * 676 + ky * 26 + kx;
            bv0[h] = Xb0[off];                  // halo => unconditional
            bv1[h] = Xb1[off];
        }
        __builtin_prefetch(A0 + k0 + 64, 0, 3);
        __builtin_prefetch(A1 + k0 + 64, 0, 3);
        acc00 = __builtin_amdgcn_wmma_f32_16x16x32_f16(
                    false, av0, false, bv0, (short)0, acc00, false, false);
        acc01 = __builtin_amdgcn_wmma_f32_16x16x32_f16(
                    false, av0, false, bv1, (short)0, acc01, false, false);
        acc10 = __builtin_amdgcn_wmma_f32_16x16x32_f16(
                    false, av1, false, bv0, (short)0, acc10, false, false);
        acc11 = __builtin_amdgcn_wmma_f32_16x16x32_f16(
                    false, av1, false, bv1, (short)0, acc11, false, false);
    }

    _Float16* Yb = Yp + (long)bb * Cout * 676;
    #pragma unroll
    for (int q = 0; q < 4; ++q) {
        const v8f& acc = (q == 0) ? acc00 : (q == 1) ? acc01
                       : (q == 2) ? acc10 : acc11;
        const int mbase = m0 + ((q >> 1) << 4);
        const int nbase = n0 + ((q & 1) << 4);
        const int ccol  = nbase + (lane & 15);
        if (ccol < 484) {
            const int cy = ccol / 22, cx = ccol - cy * 22;
            const int rbase = mbase + ((lane >> 4) << 3);
            #pragma unroll
            for (int r = 0; r < 8; ++r) {
                int row = rbase + r;
                float v = fmaxf(acc[r] + bias[row], 0.f);
                Yb[((long)row * 26 + cy + 2) * 26 + (cx + 2)] = (_Float16)v;
            }
        }
    }
}

// ---------------------------------------------------------------------------
// Conversion helpers
// ---------------------------------------------------------------------------
__global__ void cvt_rows_kernel(const float* __restrict__ src, int ld, long sS,
                                _Float16* __restrict__ dst, int Kp, long sD,
                                int rows, int K)
{
    long t = (long)blockIdx.x * blockDim.x + threadIdx.x;
    if (t >= (long)rows * Kp) return;
    int r = (int)(t / Kp), k = (int)(t - (long)r * Kp);
    float v = (k < K) ? src[(long)blockIdx.z * sS + (long)r * ld + k] : 0.f;
    dst[(long)blockIdx.z * sD + t] = (_Float16)v;
}

__global__ void cvtT_kernel(const float* __restrict__ src, int ld, long sS,
                            _Float16* __restrict__ dst, int Kp, long sD,
                            int K, int N)
{
    long t = (long)blockIdx.x * blockDim.x + threadIdx.x;
    if (t >= (long)N * Kp) return;
    int n = (int)(t / Kp), k = (int)(t - (long)n * Kp);
    float v = (k < K) ? src[(long)blockIdx.z * sS + (long)k * ld + n] : 0.f;
    dst[(long)blockIdx.z * sD + t] = (_Float16)v;
}

__global__ void wcatT_kernel(const float* __restrict__ W_rel,
                             const float* __restrict__ W_self,
                             _Float16* __restrict__ WcatT /* (512,2688) */)
{
    long t = (long)blockIdx.x * blockDim.x + threadIdx.x;
    if (t >= 512L * 2688) return;
    int n = (int)(t / 2688), k = (int)(t - (long)n * 2688);
    float v = 0.f;
    if (k < 2128)      v = W_rel[(long)k * 512 + n];
    else if (k < 2660) v = W_self[(long)(k - 2128) * 512 + n];
    WcatT[t] = (_Float16)v;
}

__global__ void zero_h_kernel(_Float16* __restrict__ p, long n)
{
    long t = (long)blockIdx.x * blockDim.x + threadIdx.x;
    if (t < n) p[t] = (_Float16)0.f;
}

// ---------------------------------------------------------------------------
// link_nodes[b,l,d] = sum_j colmean[j] * seq[b, ls+j, d]  (att_mean fused away)
// ---------------------------------------------------------------------------
__global__ void link_nodes_kernel(const float* __restrict__ att,
                                  const float* __restrict__ seq,
                                  const int* __restrict__ link_start,
                                  float* __restrict__ out /* (B,30,512) */)
{
    __shared__ float colmean[16];
    const int b = blockIdx.x / 30, l = blockIdx.x % 30;
    const int ls = link_start[b * 30 + l];
    if (threadIdx.x < 16) {
        int j = threadIdx.x;
        float s = 0.f;
        for (int hd = 0; hd < 12; ++hd)
            for (int i = 0; i < 16; ++i)
                s += att[((long)(b * 12 + hd) * 1024 + (ls + i)) * 1024 + (ls + j)];
        colmean[j] = s * (1.f / (12.f * 16.f));
    }
    __syncthreads();
    for (int d = threadIdx.x; d < 512; d += blockDim.x) {
        float s = 0.f;
        #pragma unroll
        for (int j = 0; j < 16; ++j)
            s += colmean[j] * seq[((long)b * 1024 + (ls + j)) * 512 + d];
        out[((long)b * 30 + l) * 512 + d] = s;
    }
}

__global__ void mention_kernel(const float* __restrict__ seq,
                               const int* __restrict__ midx,
                               float* __restrict__ m_emb,
                               float* __restrict__ ent_nodes)
{
    const int b = blockIdx.x / 22, e = blockIdx.x % 22;
    const int i0 = midx[(b * 22 + e) * 3 + 0];
    const int i1 = midx[(b * 22 + e) * 3 + 1];
    const int i2 = midx[(b * 22 + e) * 3 + 2];
    for (int d = threadIdx.x; d < 512; d += blockDim.x) {
        float s0 = seq[((long)b * 1024 + i0) * 512 + d];
        float s1 = seq[((long)b * 1024 + i1) * 512 + d];
        float s2 = seq[((long)b * 1024 + i2) * 512 + d];
        m_emb[((long)b * 66 + e * 3 + 0) * 512 + d] = s0;
        m_emb[((long)b * 66 + e * 3 + 1) * 512 + d] = s1;
        m_emb[((long)b * 66 + e * 3 + 2) * 512 + d] = s2;
        float mx = fmaxf(s0, fmaxf(s1, s2));
        ent_nodes[((long)b * 22 + e) * 512 + d] =
            mx + logf(expf(s0 - mx) + expf(s1 - mx) + expf(s2 - mx));
    }
}

__global__ void ea_kernel(const float* __restrict__ att,
                          const int* __restrict__ midx,
                          _Float16* __restrict__ eaH /* (B*22,1024) */)
{
    __shared__ float red[256];
    const int b = blockIdx.x / 22, e = blockIdx.x % 22;
    int idx0 = midx[(b * 22 + e) * 3 + 0];
    int idx1 = midx[(b * 22 + e) * 3 + 1];
    int idx2 = midx[(b * 22 + e) * 3 + 2];
    float v[4]; float loc = 0.f;
    for (int q = 0; q < 4; ++q) {
        int j = threadIdx.x + q * 256;
        float s = 0.f;
        for (int hd = 0; hd < 12; ++hd) {
            const float* base = att + (long)(b * 12 + hd) * 1024 * 1024 + j;
            s += base[(long)idx0 * 1024] + base[(long)idx1 * 1024]
               + base[(long)idx2 * 1024];
        }
        v[q] = s * (1.f / 36.f);
        loc += v[q];
    }
    red[threadIdx.x] = loc; __syncthreads();
    for (int off = 128; off; off >>= 1) {
        if (threadIdx.x < off) red[threadIdx.x] += red[threadIdx.x + off];
        __syncthreads();
    }
    float inv = 1.f / (red[0] + 1e-5f);
    for (int q = 0; q < 4; ++q)
        eaH[((long)b * 22 + e) * 1024 + threadIdx.x + q * 256] =
            (_Float16)(v[q] * inv);
}

__global__ void nodes_kernel(const float* __restrict__ ent_nodes,
                             const float* __restrict__ m_emb,
                             const float* __restrict__ link_nodes,
                             const float* __restrict__ type_embed,
                             const int* __restrict__ node_types,
                             _Float16* __restrict__ AcatH /* (472,2688) */)
{
    const int b = blockIdx.x / 118, n = blockIdx.x % 118;
    _Float16* dst = AcatH + ((long)b * 118 + n) * 2688 + 2128;
    for (int d = threadIdx.x; d < 560; d += blockDim.x) {
        float v = 0.f;
        if (d < 512) {
            if (n < 22)      v = ent_nodes[((long)b * 22 + n) * 512 + d];
            else if (n < 88) v = m_emb[((long)b * 66 + (n - 22)) * 512 + d];
            else             v = link_nodes[((long)b * 30 + (n - 88)) * 512 + d];
        } else if (d < 532) {
            v = type_embed[node_types[b * 118 + n] * 20 + (d - 512)];
        }
        dst[d] = (_Float16)v;
    }
}

__global__ void msg_kernel(const float* __restrict__ adj,
                           _Float16* __restrict__ AcatH)
{
    __shared__ float red[256];
    const int i = blockIdx.x % 118;
    const int r = (blockIdx.x / 118) % 4;
    const int b = blockIdx.x / (118 * 4);
    const float* arow = adj + ((long)(b * 4 + r) * 118 + i) * 118;
    float loc = 0.f;
    for (int j = threadIdx.x; j < 118; j += blockDim.x) loc += arow[j];
    red[threadIdx.x] = loc; __syncthreads();
    for (int off = 128; off; off >>= 1) {
        if (threadIdx.x < off) red[threadIdx.x] += red[threadIdx.x + off];
        __syncthreads();
    }
    const float inv = 1.f / (red[0] + 1e-5f);
    const _Float16* nodes = AcatH + (long)b * 118 * 2688 + 2128;
    _Float16* dst = AcatH + ((long)b * 118 + i) * 2688 + r * 532;
    for (int d = threadIdx.x; d < 532; d += blockDim.x) {
        float s = 0.f;
        for (int j = 0; j < 118; ++j)
            s += arow[j] * (float)nodes[(long)j * 2688 + d];
        dst[d] = (_Float16)(s * inv);
    }
}

__global__ void xinit_kernel(const float* __restrict__ gcn,
                             const float* __restrict__ ectx,
                             _Float16* __restrict__ Xp0 /* (B,512,26,26) */)
{
    long t = (long)blockIdx.x * blockDim.x + threadIdx.x;
    if (t >= 4L * 512 * 676) return;
    int ix = (int)(t % 26); long r = t / 26;
    int iy = (int)(r % 26); r /= 26;
    int d  = (int)(r % 512);
    int b  = (int)(r / 512);
    float v = 0.f;
    if (iy >= 2 && iy < 24 && ix >= 2 && ix < 24) {
        int i = iy - 2, k = ix - 2;
        float e1 = gcn[((long)b * 118 + i) * 512 + d];
        float e2 = gcn[((long)b * 118 + k) * 512 + d];
        float c1 = ectx[((long)b * 22 + i) * 512 + d];
        float c2 = ectx[((long)b * 22 + k) * 512 + d];
        v = e1 * e2 + c1 * c2;
    }
    Xp0[t] = (_Float16)v;
}

__global__ void feat_kernel(const float* __restrict__ gcn,
                            const _Float16* __restrict__ x3p,
                            const int* __restrict__ hts,
                            _Float16* __restrict__ featH /* (B*462,2048) */)
{
    const int b = blockIdx.x / 462, p = blockIdx.x % 462;
    const int hi = hts[(b * 462 + p) * 2 + 0];
    const int ti = hts[(b * 462 + p) * 2 + 1];
    _Float16* f = featH + (long)(b * 462 + p) * 2048;
    const float* eh = gcn + ((long)b * 118 + hi) * 512;
    const float* et = gcn + ((long)b * 118 + ti) * 512;
    for (int d = threadIdx.x; d < 512; d += blockDim.x) {
        float hv = eh[d], tv = et[d];
        float mv = (float)x3p[(((long)b * 512 + d) * 26 + hi + 2) * 26 + (ti + 2)];
        f[d]        = (_Float16)hv;
        f[512 + d]  = (_Float16)tv;
        f[1024 + d] = (_Float16)mv;
        f[1536 + d] = (_Float16)(hv * tv);
    }
}

// ---------------------------------------------------------------------------
static inline void launch_gemm(const _Float16* Ah, int Kp, long sA,
                               const _Float16* BT, long sB,
                               float* C, int ldc, long sC,
                               _Float16* Ch, int ldch, long sCh,
                               int M, int N,
                               const float* bias, int act, int batch,
                               hipStream_t s)
{
    int bm = (M + 31) / 32, bn = (N + 31) / 32;
    int blocks = (bm * bn + 3) / 4;
    wmma_gemm_f16<<<dim3(blocks, 1, batch), 128, 0, s>>>(
        Ah, Kp, sA, BT, sB, C, ldc, sC, Ch, ldch, sCh, M, N, bias, act);
}

static inline void launch_conv(const _Float16* Xp, const _Float16* Wh,
                               const float* bias, _Float16* Yp,
                               int Cin, int Cout, hipStream_t s)
{
    int blocks = ((Cout / 32) * 16 + 3) / 4;
    wmma_conv_f16<<<dim3(blocks, 1, 4), 128, 0, s>>>(Xp, Wh, bias, Yp, Cin, Cout);
}

static inline int cdiv(long n, int b) { return (int)((n + b - 1) / b); }

extern "C" void kernel_launch(void* const* d_in, const int* in_sizes, int n_in,
                              void* d_out, int out_size, void* d_ws, size_t ws_size,
                              hipStream_t stream)
{
    (void)in_sizes; (void)n_in; (void)out_size; (void)ws_size;

    const float* seq_out     = (const float*)d_in[0];
    const float* att         = (const float*)d_in[1];
    const float* adj         = (const float*)d_in[2];
    const int*   mention_idx = (const int*)  d_in[3];
    const int*   link_start  = (const int*)  d_in[4];
    const int*   node_types  = (const int*)  d_in[5];
    const int*   hts         = (const int*)  d_in[6];
    const float* W_trans     = (const float*)d_in[7];
    const float* b_trans     = (const float*)d_in[8];
    const float* type_embed  = (const float*)d_in[9];
    const float* W_rel       = (const float*)d_in[10];
    const float* W_self      = (const float*)d_in[11];
    const float* b_rgcn      = (const float*)d_in[12];
    const float* conv1_w     = (const float*)d_in[13];
    const float* conv1_b     = (const float*)d_in[14];
    const float* conv2_w     = (const float*)d_in[15];
    const float* conv2_b     = (const float*)d_in[16];
    const float* conv3_w     = (const float*)d_in[17];
    const float* conv3_b     = (const float*)d_in[18];
    const float* ht_W        = (const float*)d_in[19];
    const float* ht_b        = (const float*)d_in[20];
    const float* bil_W       = (const float*)d_in[21];
    const float* bil_b       = (const float*)d_in[22];

    // ---- workspace layout; every region a multiple of 8 floats (32B) ----
    float* ws = (float*)d_ws;
    float*    seq   = ws;              ws += 4L * 1024 * 512;
    float*    lnk   = ws;              ws += 4L * 30 * 512;
    float*    memb  = ws;              ws += 4L * 66 * 512;
    float*    entn  = ws;              ws += 4L * 22 * 512;
    float*    gcn   = ws;              ws += 472L * 512;
    float*    ectx  = ws;              ws += 4L * 22 * 512;
    _Float16* seqA  = (_Float16*)ws;   ws += 4096L * 768 / 2;
    _Float16* WtT   = (_Float16*)ws;   ws += 512L * 768 / 2;
    _Float16* seqT  = (_Float16*)ws;   ws += 4L * 512 * 1024 / 2;
    _Float16* eaH   = (_Float16*)ws;   ws += 88L * 1024 / 2;
    _Float16* AcatH = (_Float16*)ws;   ws += 472L * 2688 / 2;
    _Float16* WcatT = (_Float16*)ws;   ws += 512L * 2688 / 2;
    _Float16* Xp0   = (_Float16*)ws;   ws += 4L * 512 * 676 / 2;
    _Float16* Yp1   = (_Float16*)ws;   ws += 4L * 256 * 676 / 2;
    _Float16* Yp2   = (_Float16*)ws;   ws += 4L * 256 * 676 / 2;
    _Float16* Yp3   = (_Float16*)ws;   ws += 4L * 512 * 676 / 2;
    _Float16* Wc1   = (_Float16*)ws;   ws += 256L * 12800 / 2;
    _Float16* Wc2   = (_Float16*)ws;   ws += 256L * 6400 / 2;
    _Float16* Wc3   = (_Float16*)ws;   ws += 512L * 6400 / 2;
    _Float16* featH = (_Float16*)ws;   ws += 1848L * 2048 / 2;
    _Float16* htWT  = (_Float16*)ws;   ws += 1024L * 2048 / 2;
    _Float16* htH   = (_Float16*)ws;   ws += 1848L * 1024 / 2;
    _Float16* bilWT = (_Float16*)ws;   ws += 97L * 1024 / 2 + 8;

    // 1) seq = X @ W_trans + b   (4096x768x512 WMMA)
    cvt_rows_kernel<<<cdiv(4096L * 768, 256), 256, 0, stream>>>(
        seq_out, 768, 0, seqA, 768, 0, 4096, 768);
    cvtT_kernel<<<cdiv(512L * 768, 256), 256, 0, stream>>>(
        W_trans, 512, 0, WtT, 768, 0, 768, 512);
    launch_gemm(seqA, 768, 0, WtT, 0, seq, 512, 0, nullptr, 0, 0,
                4096, 512, b_trans, 0, 1, stream);

    // 2) graph-node inputs (fused attention gathers; att_mean never built)
    link_nodes_kernel<<<120, 256, 0, stream>>>(att, seq, link_start, lnk);
    mention_kernel  <<< 88, 256, 0, stream>>>(seq, mention_idx, memb, entn);
    ea_kernel       <<< 88, 256, 0, stream>>>(att, mention_idx, eaH);
    cvtT_kernel<<<dim3(cdiv(512L * 1024, 256), 1, 4), 256, 0, stream>>>(
        seq, 512, 1024L * 512, seqT, 1024, 512L * 1024, 1024, 512);

    // 3) RGCN: f16 A_cat (K padded 2660->2688) @ stacked transposed weights
    nodes_kernel<<<472, 256, 0, stream>>>(entn, memb, lnk, type_embed,
                                          node_types, AcatH);
    msg_kernel<<<4 * 4 * 118, 256, 0, stream>>>(adj, AcatH);
    wcatT_kernel<<<cdiv(512L * 2688, 256), 256, 0, stream>>>(W_rel, W_self, WcatT);
    launch_gemm(AcatH, 2688, 0, WcatT, 0, gcn, 512, 0, nullptr, 0, 0,
                472, 512, b_rgcn, 1, 1, stream);

    // 4) e_ctx = ea @ seq (batched 22x1024x512 WMMA)
    launch_gemm(eaH, 1024, 22L * 1024, seqT, 512L * 1024,
                ectx, 512, 22L * 512, nullptr, 0, 0,
                22, 512, nullptr, 0, 4, stream);

    // 5) pairwise maps + conv stack (implicit-im2col WMMA, f16 halo tensors)
    xinit_kernel<<<cdiv(4L * 512 * 676, 256), 256, 0, stream>>>(gcn, ectx, Xp0);
    cvt_rows_kernel<<<cdiv(256L * 12800, 256), 256, 0, stream>>>(
        conv1_w, 12800, 0, Wc1, 12800, 0, 256, 12800);
    cvt_rows_kernel<<<cdiv(256L * 6400, 256), 256, 0, stream>>>(
        conv2_w, 6400, 0, Wc2, 6400, 0, 256, 6400);
    cvt_rows_kernel<<<cdiv(512L * 6400, 256), 256, 0, stream>>>(
        conv3_w, 6400, 0, Wc3, 6400, 0, 512, 6400);
    zero_h_kernel<<<cdiv(4L * 256 * 676, 256), 256, 0, stream>>>(Yp1, 4L * 256 * 676);
    zero_h_kernel<<<cdiv(4L * 256 * 676, 256), 256, 0, stream>>>(Yp2, 4L * 256 * 676);
    zero_h_kernel<<<cdiv(4L * 512 * 676, 256), 256, 0, stream>>>(Yp3, 4L * 512 * 676);
    launch_conv(Xp0, Wc1, conv1_b, Yp1, 512, 256, stream);
    launch_conv(Yp1, Wc2, conv2_b, Yp2, 256, 256, stream);
    launch_conv(Yp2, Wc3, conv3_b, Yp3, 256, 512, stream);

    // 6) pair features -> ht (tanh, f16 out) -> logits (f32 to d_out)
    feat_kernel<<<1848, 256, 0, stream>>>(gcn, Yp3, hts, featH);
    cvtT_kernel<<<cdiv(1024L * 2048, 256), 256, 0, stream>>>(
        ht_W, 1024, 0, htWT, 2048, 0, 2048, 1024);
    cvtT_kernel<<<cdiv(97L * 1024, 256), 256, 0, stream>>>(
        bil_W, 97, 0, bilWT, 1024, 0, 1024, 97);
    launch_gemm(featH, 2048, 0, htWT, 0, nullptr, 0, 0, htH, 1024, 0,
                1848, 1024, ht_b, 2, 1, stream);
    launch_gemm(htH, 1024, 0, bilWT, 0, (float*)d_out, 97, 0, nullptr, 0, 0,
                1848, 97, bil_b, 0, 1, stream);
}
